// ConvNetMinVar_57372173140200
// MI455X (gfx1250) — matrix operand order repaired and checked by reading, not compiled, not run on hardware
//
#include <hip/hip_runtime.h>
#include <hip/hip_bf16.h>
#include <stdint.h>

// ---------------- problem constants ----------------
#define NB     8192      // batch
#define NFEAT  245       // 49*5 batchnorm features
#define NL     49
#define NA     5
#define NH     120       // hidden (24*5)
#define NHP    128       // hidden padded (col 120 = 1.0 for bias fold)
#define NOUT   100       // linear outputs
#define NOBS   20
#define ADMM_ITERS   150
#define BISECT_ITERS 25
#define RHO_EPS 1e-6f

#define TILES_M (NB/16)        // 512
#define TILES_N 7              // ceil(100/16)
#define KCHUNKS (NHP/32)       // 4

typedef _Float16 v16h __attribute__((ext_vector_type(16)));
typedef float    v8f  __attribute__((ext_vector_type(8)));

// ---------------- workspace layout (bytes) ----------------
#define WS_SCALE   0                      // scale[256] f32
#define WS_SHIFT   1024                   // shift[256] f32
#define WS_CONV    2048                   // wbar[3], bbar f32
#define WS_BPACK   4096                   // u32 [7][4][8][32] = 28KB
#define WS_HBUF    32768                  // f16 [NB][NHP] = 2MB
#define WS_FEATS   (32768 + NB*NHP*2)     // f32 [NB][NOUT] = 3.2MB

// =========================================================
// Kernel 1: batchnorm batch statistics -> folded scale/shift
// =========================================================
__global__ void bn_stats_kernel(const float* __restrict__ x,
                                const float* __restrict__ gamma,
                                const float* __restrict__ beta,
                                float* __restrict__ scale,
                                float* __restrict__ shift) {
    __shared__ float s1[256];
    __shared__ float s2[256];
    const int f   = blockIdx.x;            // 0..244
    const int tid = threadIdx.x;
    float sum = 0.f, sumsq = 0.f;
    for (int b = tid; b < NB; b += 256) {
        float v = x[(size_t)b * NFEAT + f];
        sum   += v;
        sumsq += v * v;
    }
    s1[tid] = sum; s2[tid] = sumsq;
    __syncthreads();
    for (int s = 128; s > 0; s >>= 1) {
        if (tid < s) { s1[tid] += s1[tid + s]; s2[tid] += s2[tid + s]; }
        __syncthreads();
    }
    if (tid == 0) {
        float mu  = s1[0] * (1.0f / NB);
        float var = s2[0] * (1.0f / NB) - mu * mu;       // biased var
        float rs  = rsqrtf(var + 1e-5f);
        float sc  = gamma[f] * rs;
        scale[f] = sc;
        shift[f] = beta[f] - mu * sc;
    }
}

// =========================================================
// Kernel 2: fold conv (channel-mean pool => single 3-tap filter)
//           and pre-pack [lin_w ; lin_b] into WMMA B-fragment layout.
//  B frag (16-bit 32x16, wave32): VGPR v, lane L:
//    n = L&15,  k = (v>>2)*16 + (L>>4)*8 + (v&3)*2 (+0/1 packed)
//  Bias fold: weight row k==120 carries lin_b[n]; activation col 120 == 1.
// =========================================================
__global__ void prep_weights_kernel(const float* __restrict__ conv_w,  // (3,1,3,1) OIHW
                                    const float* __restrict__ conv_b,  // (3,)
                                    const float* __restrict__ lin_w,   // (100,120)
                                    const float* __restrict__ lin_b,   // (100,)
                                    float* __restrict__ convfold,      // wbar[3], bbar
                                    uint32_t* __restrict__ bpack) {
    const int tid = threadIdx.x;
    if (tid == 0) {
        const float inv3 = 1.0f / 3.0f;
        for (int kh = 0; kh < 3; ++kh)
            convfold[kh] = (conv_w[0*3+kh] + conv_w[1*3+kh] + conv_w[2*3+kh]) * inv3;
        convfold[3] = (conv_b[0] + conv_b[1] + conv_b[2]) * inv3;
    }
    const int total = TILES_N * KCHUNKS * 8 * 32;   // 7168
    for (int idx = tid; idx < total; idx += 256) {
        int lane  = idx & 31;
        int v     = (idx >> 5) & 7;
        int chunk = (idx >> 8) & 3;
        int tn    = idx >> 10;
        int n     = tn * 16 + (lane & 15);
        int k     = chunk * 32 + ((v >> 2) * 16) + ((lane >> 4) & 1) * 8 + (v & 3) * 2;
        float lo = 0.0f, hi = 0.0f;
        if (n < NOUT) {
            lo = (k     < NH) ? lin_w[n * NH + k]     : ((k     == NH) ? lin_b[n] : 0.0f);
            hi = (k + 1 < NH) ? lin_w[n * NH + k + 1] : ((k + 1 == NH) ? lin_b[n] : 0.0f);
        }
        union { _Float16 h[2]; uint32_t u; } p;
        p.h[0] = (_Float16)lo; p.h[1] = (_Float16)hi;
        bpack[idx] = p.u;
    }
}

// =========================================================
// Kernel 3: normalize + folded conv + relu -> f16 activations
//   h[r*5+a] = relu( sum_kh wbar[kh]*xn[(2r+kh)*5+a] + bbar )
//   col 120 = 1.0 (bias fold), cols 121..127 = 0
// =========================================================
__global__ void features_kernel(const float* __restrict__ x,
                                const float* __restrict__ scale,
                                const float* __restrict__ shift,
                                const float* __restrict__ convfold,
                                _Float16* __restrict__ hbuf) {
    const int s = blockIdx.x;         // sample
    const int t = threadIdx.x;        // 0..127
    _Float16 hv = (t == NH) ? (_Float16)1.0f : (_Float16)0.0f;
    if (t < NH) {
        const int r = t / NA;
        const int a = t % NA;
        const size_t base = (size_t)s * NFEAT;
        float acc = convfold[3];
        #pragma unroll
        for (int kh = 0; kh < 3; ++kh) {
            int f = (2 * r + kh) * NA + a;
            float xn = fmaf(x[base + f], scale[f], shift[f]);
            acc = fmaf(convfold[kh], xn, acc);
        }
        hv = (_Float16)fmaxf(acc, 0.0f);
    }
    hbuf[(size_t)s * NHP + t] = hv;
}

// =========================================================
// Kernel 4: WMMA GEMM  feats = [h,1](8192x121) @ [lin_w;lin_b]^T
//   one 16x16 output tile per wave, K = 4 chunks of 32 (f16)
//   A frag (16-bit 16x32): lane L -> M=L&15;
//     VGPR v -> kbase = (v>>2)*16 + ((L>>4)&1)*8 + (v&3)*2
// =========================================================
__global__ void gemm_wmma_kernel(const _Float16* __restrict__ hbuf,
                                 const uint32_t* __restrict__ bpack,
                                 float* __restrict__ feats) {
    const int lane = threadIdx.x & 31;
    const int wave = threadIdx.x >> 5;
    const int tile = blockIdx.x * 8 + wave;          // 448 blocks * 8 waves = 3584 tiles
    const int tm   = tile / TILES_N;
    const int tn   = tile % TILES_N;

    const int rowA  = tm * 16 + (lane & 15);
    const int khalf = ((lane >> 4) & 1) * 8;
    const uint32_t* hp = (const uint32_t*)(hbuf + (size_t)rowA * NHP); // 2 f16 per u32

    v8f acc = {};
    #pragma unroll
    for (int chunk = 0; chunk < KCHUNKS; ++chunk) {
        union { uint32_t u[8]; v16h v; } afrag, bfrag;
        const uint32_t* bp = bpack + ((tn * KCHUNKS + chunk) * 8) * 32;
        #pragma unroll
        for (int v = 0; v < 8; ++v) {
            int kbase = chunk * 32 + ((v >> 2) * 16) + khalf + ((v & 3) * 2);
            afrag.u[v] = hp[kbase >> 1];
            bfrag.u[v] = bp[v * 32 + lane];
        }
        acc = __builtin_amdgcn_wmma_f32_16x16x32_f16(
            /*neg_a=*/false, afrag.v, /*neg_b=*/false, bfrag.v,
            /*c_mod=*/(short)0, acc, /*reuse_a=*/false, /*reuse_b=*/false);
    }

    // C layout: VGPR j -> M = j + ((lane>=16)?8:0), N = lane&15
    const int nOut = tn * 16 + (lane & 15);
    if (nOut < NOUT) {
        const int mhalf = ((lane >> 4) & 1) * 8;
        #pragma unroll
        for (int j = 0; j < 8; ++j) {
            int row = tm * 16 + mhalf + j;
            feats[(size_t)row * NOUT + nOut] = acc[j];
        }
    }
}

// =========================================================
// Kernel 5: per-lane covariance + ADMM min-variance QP.
//   Sigma = cov_sqrt @ cov_sqrt = cov (PSD) -> eigh cancels.
// =========================================================
__global__ void admm_kernel(const float* __restrict__ feats,
                            float* __restrict__ out) {
    const int s = blockIdx.x * blockDim.x + threadIdx.x;
    if (s >= NB) return;
    const float* fp = feats + (size_t)s * NOUT;

    // mean over 20 observations
    float m[NA];
    #pragma unroll
    for (int i = 0; i < NA; ++i) m[i] = 0.0f;
    #pragma unroll 1
    for (int n = 0; n < NOBS; ++n) {
        #pragma unroll
        for (int i = 0; i < NA; ++i) m[i] += fp[n * NA + i];
    }
    #pragma unroll
    for (int i = 0; i < NA; ++i) m[i] *= (1.0f / NOBS);

    // unbiased sample covariance (upper triangle then mirror)
    float cv[NA][NA];
    #pragma unroll
    for (int i = 0; i < NA; ++i)
        #pragma unroll
        for (int j = 0; j < NA; ++j) cv[i][j] = 0.0f;
    #pragma unroll 1
    for (int n = 0; n < NOBS; ++n) {
        float c[NA];
        #pragma unroll
        for (int i = 0; i < NA; ++i) c[i] = fp[n * NA + i] - m[i];
        #pragma unroll
        for (int i = 0; i < NA; ++i)
            #pragma unroll
            for (int j = i; j < NA; ++j) cv[i][j] = fmaf(c[i], c[j], cv[i][j]);
    }
    const float invnm1 = 1.0f / (NOBS - 1);
    #pragma unroll
    for (int i = 0; i < NA; ++i)
        #pragma unroll
        for (int j = i; j < NA; ++j) { cv[i][j] *= invnm1; cv[j][i] = cv[i][j]; }

    // rho = trace(Sigma)/n + eps ; M = 2*Sigma + rho*I ; Minv by Gauss-Jordan (PD)
    float rho = RHO_EPS;
    #pragma unroll
    for (int i = 0; i < NA; ++i) rho += cv[i][i] * (1.0f / NA);

    float A[NA][NA], Inv[NA][NA];
    #pragma unroll
    for (int i = 0; i < NA; ++i)
        #pragma unroll
        for (int j = 0; j < NA; ++j) {
            A[i][j]   = 2.0f * cv[i][j] + ((i == j) ? rho : 0.0f);
            Inv[i][j] = (i == j) ? 1.0f : 0.0f;
        }
    #pragma unroll
    for (int k = 0; k < NA; ++k) {
        float d = 1.0f / A[k][k];
        #pragma unroll
        for (int j = 0; j < NA; ++j) { A[k][j] *= d; Inv[k][j] *= d; }
        #pragma unroll
        for (int i = 0; i < NA; ++i) {
            if (i == k) continue;
            float f = A[i][k];
            #pragma unroll
            for (int j = 0; j < NA; ++j) {
                A[i][j]   = fmaf(-f, A[k][j],   A[i][j]);
                Inv[i][j] = fmaf(-f, Inv[k][j], Inv[i][j]);
            }
        }
    }

    // ADMM
    float z[NA], u[NA];
    #pragma unroll
    for (int i = 0; i < NA; ++i) { z[i] = 1.0f / NA; u[i] = 0.0f; }

    #pragma unroll 1
    for (int it = 0; it < ADMM_ITERS; ++it) {
        float t[NA], w[NA], v[NA];
        #pragma unroll
        for (int i = 0; i < NA; ++i) t[i] = rho * (z[i] - u[i]);
        #pragma unroll
        for (int i = 0; i < NA; ++i) {
            float acc = 0.0f;
            #pragma unroll
            for (int j = 0; j < NA; ++j) acc = fmaf(Inv[i][j], t[j], acc);
            w[i] = acc;
            v[i] = acc + u[i];
        }
        // projection onto {sum=1, -1<=z<=1}: bisection + one Newton step
        float lo = v[0], hi = v[0];
        #pragma unroll
        for (int i = 1; i < NA; ++i) { lo = fminf(lo, v[i]); hi = fmaxf(hi, v[i]); }
        lo -= 2.0f; hi += 2.0f;
        #pragma unroll 1
        for (int b = 0; b < BISECT_ITERS; ++b) {
            float mid = 0.5f * (lo + hi);
            float ssum = 0.0f;
            #pragma unroll
            for (int i = 0; i < NA; ++i)
                ssum += fminf(fmaxf(v[i] - mid, -1.0f), 1.0f);
            bool big = ssum > 1.0f;
            lo = big ? mid : lo;
            hi = big ? hi : mid;
        }
        float tau0 = 0.5f * (lo + hi);
        float cnt = 0.0f, g = 0.0f;
        #pragma unroll
        for (int i = 0; i < NA; ++i) {
            float w0 = v[i] - tau0;
            cnt += (w0 > -1.0f && w0 < 1.0f) ? 1.0f : 0.0f;
            g   += fminf(fmaxf(w0, -1.0f), 1.0f);
        }
        cnt = fmaxf(cnt, 1.0f);
        float tau = tau0 + (g - 1.0f) / cnt;
        #pragma unroll
        for (int i = 0; i < NA; ++i) {
            float zn = fminf(fmaxf(v[i] - tau, -1.0f), 1.0f);
            u[i] = u[i] + w[i] - zn;
            z[i] = zn;
        }
    }

    #pragma unroll
    for (int i = 0; i < NA; ++i) out[(size_t)s * NA + i] = z[i];
}

// =========================================================
extern "C" void kernel_launch(void* const* d_in, const int* in_sizes, int n_in,
                              void* d_out, int out_size, void* d_ws, size_t ws_size,
                              hipStream_t stream) {
    const float* x      = (const float*)d_in[0];   // (8192,1,49,5)
    const float* gamma  = (const float*)d_in[1];   // (245,)
    const float* beta   = (const float*)d_in[2];   // (245,)
    const float* conv_w = (const float*)d_in[3];   // (3,1,3,1)
    const float* conv_b = (const float*)d_in[4];   // (3,)
    const float* lin_w  = (const float*)d_in[5];   // (100,120)
    const float* lin_b  = (const float*)d_in[6];   // (100,)

    char* ws = (char*)d_ws;
    float*    scale    = (float*)(ws + WS_SCALE);
    float*    shift    = (float*)(ws + WS_SHIFT);
    float*    convfold = (float*)(ws + WS_CONV);
    uint32_t* bpack    = (uint32_t*)(ws + WS_BPACK);
    _Float16* hbuf     = (_Float16*)(ws + WS_HBUF);
    float*    feats    = (float*)(ws + WS_FEATS);

    bn_stats_kernel<<<NFEAT, 256, 0, stream>>>(x, gamma, beta, scale, shift);
    prep_weights_kernel<<<1, 256, 0, stream>>>(conv_w, conv_b, lin_w, lin_b,
                                               convfold, bpack);
    features_kernel<<<NB, NHP, 0, stream>>>(x, scale, shift, convfold, hbuf);
    gemm_wmma_kernel<<<(TILES_M * TILES_N) / 8, 256, 0, stream>>>(hbuf, bpack, feats);
    admm_kernel<<<NB / 256, 256, 0, stream>>>(feats, (float*)d_out);
}
